// QuantumLogicCore_23433341567228
// MI455X (gfx1250) — matrix operand order repaired and checked by reading, not compiled, not run on hardware
//
#include <hip/hip_runtime.h>

#define DDIM   512
#define KDIM   1024      // 2*d flattened re/im
#define HN     4
#define MBANK  2048
#define TOPK   4
#define RNK    8
#define NTOK   4096      // B*T
#define TMAXI  4
#define EPSF   1e-6f
#define THRESHF 0.99f

typedef __attribute__((ext_vector_type(16))) __bf16          bf16x16;
typedef __attribute__((ext_vector_type(8)))  float           f32x8;
typedef __attribute__((ext_vector_type(16))) unsigned short  u16x16;
typedef __attribute__((ext_vector_type(8)))  unsigned short  u16x8;
typedef __attribute__((ext_vector_type(4)))  unsigned int    u32x4;
typedef __attribute__((ext_vector_type(8)))  int             i32x8;
typedef __attribute__((ext_vector_type(4)))  int             i32x4;

#if defined(__has_builtin)
#  if __has_builtin(__builtin_amdgcn_tensor_load_to_lds)
#    define HAVE_TDM 1
#  endif
#  if __has_builtin(__builtin_amdgcn_s_wait_tensorcnt)
#    define HAVE_WAIT_TENSOR 1
#  endif
#endif

__device__ __forceinline__ unsigned short f2bf(float f) {
  unsigned int u = __builtin_bit_cast(unsigned int, f);
  u += 0x7FFFu + ((u >> 16) & 1u);
  return (unsigned short)(u >> 16);
}

__device__ __forceinline__ float wsum(float v) {
  v += __shfl_xor(v, 16); v += __shfl_xor(v, 8); v += __shfl_xor(v, 4);
  v += __shfl_xor(v, 2);  v += __shfl_xor(v, 1);
  return v;
}

// ---------------------------------------------------------------------------
// init: bank_keys fp32 -> bf16  [H*M][1024]
// ---------------------------------------------------------------------------
__global__ void __launch_bounds__(256)
k_init_keys(const float* __restrict__ keys, unsigned short* __restrict__ keysb, int count) {
  int i = blockIdx.x * 256 + threadIdx.x;
  if (i < count) keysb[i] = f2bf(keys[i]);
}

// ---------------------------------------------------------------------------
// init: psi_h broadcast, pool bf16, zero ACT state + merged + cost scalar
// ---------------------------------------------------------------------------
__global__ void __launch_bounds__(256)
k_init_state(const float* __restrict__ psi, float* __restrict__ psi_h,
             unsigned short* __restrict__ pool, float* __restrict__ merged,
             float* __restrict__ cum, float* __restrict__ cost,
             int* __restrict__ halted, float* __restrict__ cost_out) {
  int i = blockIdx.x * 256 + threadIdx.x;
  if (i < NTOK * KDIM) {
    float v = psi[i];
    int n = i >> 10, j = i & (KDIM - 1);
#pragma unroll
    for (int h = 0; h < HN; ++h)
      psi_h[(size_t)(n * HN + h) * KDIM + j] = v;
    pool[i] = f2bf(v);
    merged[i] = 0.f;
  }
  if (i < NTOK * HN) { cum[i] = 0.f; cost[i] = 0.f; halted[i] = 0; }
  if (i == 0) *cost_out = 0.f;
}

// ---------------------------------------------------------------------------
// probe + top-k. Block = 16 tokens, 256 threads (8 waves).
// dynamic LDS: A-tile bf16 [16][1024] (TDM-loaded) + logits f32 [16][2048]
// ---------------------------------------------------------------------------
#define A_BYTES (16 * KDIM * 2)
#define PROBE_LDS (A_BYTES + 16 * MBANK * 4)

__global__ void __launch_bounds__(256)
k_probe_topk(const unsigned short* __restrict__ pool,   // [N][1024] bf16
             const unsigned short* __restrict__ keysb,  // [H*M][1024] bf16
             int* __restrict__ sel_idx,                 // [N][H][4]
             float* __restrict__ sel_w) {               // [N][H][4]
  extern __shared__ char smem[];
  unsigned short* As = (unsigned short*)smem;
  float* Ls = (float*)(smem + A_BYTES);

  const int row0 = blockIdx.x * 16;
  const int tid  = threadIdx.x;
  const int lane = tid & 31, wave = tid >> 5;
  const int mrow = lane & 15, hi = lane >> 4;

#if defined(HAVE_TDM)
  // ---- stage A tile via Tensor Data Mover: 2D tile 1024x16 of 2-byte elems
  if (wave == 0) {
    unsigned long long gaddr = (unsigned long long)(const void*)(pool + (size_t)row0 * KDIM);
    unsigned int lds_off = (unsigned int)(unsigned long long)(void*)As;  // low 32 bits = LDS byte offset
    u32x4 g0;
    g0[0] = 1u;                                          // count=1, no gather
    g0[1] = lds_off;                                     // lds_addr
    g0[2] = (unsigned int)(gaddr & 0xFFFFFFFFull);       // global_addr[31:0]
    g0[3] = (unsigned int)((gaddr >> 32) & 0x1FFFFFFull) // global_addr[56:32]
          | (2u << 30);                                  // type=2 (image)
    i32x8 g1;
    g1[0] = (int)(1u << 16);                 // data_size=1 (2B), mask=0
    g1[1] = (int)((KDIM & 0xFFFF) << 16);    // tensor_dim0[15:0]=1024 (bits 63:48)
    g1[2] = (int)((NTOK & 0xFFFF) << 16);    // tensor_dim0 hi=0 | tensor_dim1 lo=4096
    g1[3] = (int)((KDIM & 0xFFFF) << 16);    // tensor_dim1 hi=0 | tile_dim0=1024
    g1[4] = 16;                              // tile_dim1=16, tile_dim2=0
    g1[5] = KDIM;                            // tensor_dim0_stride=1024
    g1[6] = 0;                               // stride hi / dim1_stride lo
    g1[7] = 0;
    i32x4 gz = {0, 0, 0, 0};
#if __clang_major__ >= 23
    i32x8 gz8 = {0, 0, 0, 0, 0, 0, 0, 0};
    __builtin_amdgcn_tensor_load_to_lds(g0, g1, gz, gz, gz8, 0);
#else
    __builtin_amdgcn_tensor_load_to_lds(g0, g1, gz, gz, 0);
#endif
#if defined(HAVE_WAIT_TENSOR)
    __builtin_amdgcn_s_wait_tensorcnt(0);
#else
    asm volatile("s_wait_tensorcnt 0x0" ::: "memory");
#endif
  }
#else
  // fallback: manual staging
  for (int i = tid; i < 16 * KDIM; i += 256) {
    int r = i >> 10, c = i & (KDIM - 1);
    As[r * KDIM + c] = pool[(size_t)(row0 + r) * KDIM + c];
  }
#endif
  __syncthreads();

  for (int h = 0; h < HN; ++h) {
    // ---- GEMM: logits[16 tokens][2048 cols]; 4 column tiles per wave pass
    for (int gi = 0; gi < 4; ++gi) {
      const int ct0 = wave * 4 + gi * 32;               // first of 4 tiles
      const unsigned short* brow[4];
#pragma unroll
      for (int j = 0; j < 4; ++j)
        brow[j] = keysb + (size_t)(h * MBANK + (ct0 + j) * 16 + mrow) * KDIM;
      __builtin_prefetch(brow[0] + KDIM, 0, 0);         // global_prefetch
      f32x8 acc0 = {0.f,0.f,0.f,0.f,0.f,0.f,0.f,0.f};
      f32x8 acc1 = acc0, acc2 = acc0, acc3 = acc0;
      for (int kk = 0; kk < KDIM / 32; ++kk) {
        // A fragment (row M = lane&15, K = hi*16 .. hi*16+15 contiguous, as B) — shared by 4 tiles
        const unsigned short* ap0 = &As[mrow * KDIM + kk * 32 + hi * 16];
        u16x8 a0 = *(const u16x8*)ap0;
        u16x8 a1 = *(const u16x8*)(ap0 + 8);
        u16x16 au;
#pragma unroll
        for (int j = 0; j < 8; ++j) { au[j] = a0[j]; au[8 + j] = a1[j]; }
        bf16x16 a = __builtin_bit_cast(bf16x16, au);
        // B fragments: lane = col N, K = hi*16 .. hi*16+15 contiguous
        bf16x16 b0 = __builtin_bit_cast(bf16x16, *(const u16x16*)(brow[0] + kk * 32 + hi * 16));
        bf16x16 b1 = __builtin_bit_cast(bf16x16, *(const u16x16*)(brow[1] + kk * 32 + hi * 16));
        bf16x16 b2 = __builtin_bit_cast(bf16x16, *(const u16x16*)(brow[2] + kk * 32 + hi * 16));
        bf16x16 b3 = __builtin_bit_cast(bf16x16, *(const u16x16*)(brow[3] + kk * 32 + hi * 16));
        acc0 = __builtin_amdgcn_wmma_f32_16x16x32_bf16(false, a, false, b0, (short)0, acc0, false, false);
        acc1 = __builtin_amdgcn_wmma_f32_16x16x32_bf16(false, a, false, b1, (short)0, acc1, false, false);
        acc2 = __builtin_amdgcn_wmma_f32_16x16x32_bf16(false, a, false, b2, (short)0, acc2, false, false);
        acc3 = __builtin_amdgcn_wmma_f32_16x16x32_bf16(false, a, false, b3, (short)0, acc3, false, false);
      }
      // C layout: VGPR v -> M = v + 8*hi, N = lane&15
#pragma unroll
      for (int v = 0; v < 8; ++v) {
        Ls[(v + 8 * hi) * MBANK + (ct0 + 0) * 16 + mrow] = acc0[v];
        Ls[(v + 8 * hi) * MBANK + (ct0 + 1) * 16 + mrow] = acc1[v];
        Ls[(v + 8 * hi) * MBANK + (ct0 + 2) * 16 + mrow] = acc2[v];
        Ls[(v + 8 * hi) * MBANK + (ct0 + 3) * 16 + mrow] = acc3[v];
      }
    }
    __syncthreads();

    // ---- top-4 + softmax; wave w handles tokens 2w, 2w+1 ----
    for (int tt = 0; tt < 2; ++tt) {
      const int tok = wave * 2 + tt;
      const float* lrow = &Ls[tok * MBANK];
      int   sel[4] = {-1, -1, -1, -1};
      float val[4];
#pragma unroll
      for (int p = 0; p < 4; ++p) {
        float bv = -3.4e38f; int bi = 0x7FFFFFFF;
        for (int c = lane; c < MBANK; c += 32) {
          if (c == sel[0] || c == sel[1] || c == sel[2]) continue;
          float x = lrow[c];
          if (x > bv || (x == bv && c < bi)) { bv = x; bi = c; }
        }
        for (int off = 16; off; off >>= 1) {
          float ov = __shfl_xor(bv, off);
          int   oi = __shfl_xor(bi, off);
          if (ov > bv || (ov == bv && oi < bi)) { bv = ov; bi = oi; }
        }
        sel[p] = bi; val[p] = bv;
      }
      float mx = val[0];
      float e0 = __expf(val[0] - mx), e1 = __expf(val[1] - mx);
      float e2 = __expf(val[2] - mx), e3 = __expf(val[3] - mx);
      float inv = 1.f / (e0 + e1 + e2 + e3);
      if (lane == 0) {
        int base = ((row0 + tok) * HN + h) * TOPK;
        sel_idx[base + 0] = sel[0]; sel_idx[base + 1] = sel[1];
        sel_idx[base + 2] = sel[2]; sel_idx[base + 3] = sel[3];
        sel_w[base + 0] = e0 * inv; sel_w[base + 1] = e1 * inv;
        sel_w[base + 2] = e2 * inv; sel_w[base + 3] = e3 * inv;
      }
    }
    __syncthreads();
  }
}

// ---------------------------------------------------------------------------
// projector apply: one wave per (n,h); psi_h updated in place
// ---------------------------------------------------------------------------
__global__ void __launch_bounds__(256)
k_apply(float* __restrict__ psi_h,
        const float* __restrict__ bank_u,   // [H][M][512][2]
        const float* __restrict__ bank_v,
        const float* __restrict__ mix_u,    // [H][8][4]
        const float* __restrict__ mix_v,
        const int* __restrict__ sel_idx,
        const float* __restrict__ sel_w) {
  const int gw = blockIdx.x * 8 + (threadIdx.x >> 5);  // (n,h) wave
  const int lane = threadIdx.x & 31;
  const int n = gw >> 2, h = gw & 3;
  float* psi = psi_h + (size_t)(n * HN + h) * KDIM;

  int   idx[4]; float w[4];
#pragma unroll
  for (int k = 0; k < 4; ++k) {
    idx[k] = sel_idx[(n * HN + h) * 4 + k];
    w[k]   = sel_w[(n * HN + h) * 4 + k];
  }
  const float* su[4]; const float* sv[4];
#pragma unroll
  for (int k = 0; k < 4; ++k) {
    su[k] = bank_u + (size_t)(h * MBANK + idx[k]) * KDIM;
    sv[k] = bank_v + (size_t)(h * MBANK + idx[k]) * KDIM;
  }

  float srr[4] = {0,0,0,0}, sri[4] = {0,0,0,0};
  float gu[10] = {0,0,0,0,0,0,0,0,0,0}, gv[10] = {0,0,0,0,0,0,0,0,0,0};
  float hu[6]  = {0,0,0,0,0,0};

  // ---- pass 1: dots + Grams over d (lane handles 16 complex dims) ----
  for (int t = 0; t < DDIM / 32; ++t) {
    int dd = lane + 32 * t;
    float pr = psi[2 * dd], pi = psi[2 * dd + 1];
    float ur[4], ui[4], vr[4], vi[4];
#pragma unroll
    for (int k = 0; k < 4; ++k) {
      ur[k] = su[k][2 * dd]; ui[k] = su[k][2 * dd + 1];
      vr[k] = sv[k][2 * dd]; vi[k] = sv[k][2 * dd + 1];
    }
#pragma unroll
    for (int k = 0; k < 4; ++k) {
      srr[k] += vr[k] * pr + vi[k] * pi;
      sri[k] += vr[k] * pi - vi[k] * pr;
    }
    int q = 0;
#pragma unroll
    for (int k = 0; k < 4; ++k)
#pragma unroll
      for (int k2 = k; k2 < 4; ++k2) {
        gu[q] += ur[k] * ur[k2] + ui[k] * ui[k2];
        gv[q] += vr[k] * vr[k2] + vi[k] * vi[k2];
        ++q;
      }
    q = 0;
#pragma unroll
    for (int k = 0; k < 4; ++k)
#pragma unroll
      for (int k2 = k + 1; k2 < 4; ++k2) {
        hu[q] += ur[k] * ui[k2] - ui[k] * ur[k2];
        ++q;
      }
  }
#pragma unroll
  for (int k = 0; k < 4; ++k) { srr[k] = wsum(srr[k]); sri[k] = wsum(sri[k]); }
#pragma unroll
  for (int q = 0; q < 10; ++q) { gu[q] = wsum(gu[q]); gv[q] = wsum(gv[q]); }
#pragma unroll
  for (int q = 0; q < 6; ++q) hu[q] = wsum(hu[q]);

  // ---- expand Grams ----
  float Gf[4][4], Vf[4][4], Hf[4][4];
  {
    int q = 0;
#pragma unroll
    for (int k = 0; k < 4; ++k)
#pragma unroll
      for (int k2 = k; k2 < 4; ++k2) {
        Gf[k][k2] = gu[q]; Gf[k2][k] = gu[q];
        Vf[k][k2] = gv[q]; Vf[k2][k] = gv[q]; ++q;
      }
    q = 0;
#pragma unroll
    for (int k = 0; k < 4; ++k) {
      Hf[k][k] = 0.f;
#pragma unroll
      for (int k2 = k + 1; k2 < 4; ++k2) { Hf[k][k2] = hu[q]; Hf[k2][k] = -hu[q]; ++q; }
    }
  }

  // ---- rank-8 algebra: c = normalized V^H psi; beta = U-mix of c ----
  float bre[4] = {0,0,0,0}, bim[4] = {0,0,0,0};
#pragma unroll
  for (int r = 0; r < RNK; ++r) {
    float cuk[4], cvk[4];
#pragma unroll
    for (int k = 0; k < 4; ++k) {
      cuk[k] = mix_u[(h * RNK + r) * 4 + k] * w[k];
      cvk[k] = mix_v[(h * RNK + r) * 4 + k] * w[k];
    }
    float nu = 0.f, nv = 0.f;
#pragma unroll
    for (int k = 0; k < 4; ++k)
#pragma unroll
      for (int k2 = 0; k2 < 4; ++k2) {
        nu += cuk[k] * cuk[k2] * Gf[k][k2];
        nv += cvk[k] * cvk[k2] * Vf[k][k2];
      }
    float inu = 1.f / sqrtf(fmaxf(nu, EPSF));
    float inv = 1.f / sqrtf(fmaxf(nv, EPSF));
    float a = 0.f, b = 0.f;
#pragma unroll
    for (int k = 0; k < 4; ++k) { a += cvk[k] * srr[k]; b += cvk[k] * sri[k]; }
    float cre = a * inv, cim = b * inv;
#pragma unroll
    for (int k = 0; k < 4; ++k) {
      bre[k] += cuk[k] * inu * cre;
      bim[k] += cuk[k] * inu * cim;
    }
  }

  // ---- ||o||^2 from Gram (no extra pass over o) ----
  float ss = 0.f;
#pragma unroll
  for (int k = 0; k < 4; ++k)
#pragma unroll
    for (int k2 = 0; k2 < 4; ++k2)
      ss += (bre[k] * bre[k2] + bim[k] * bim[k2]) * Gf[k][k2]
          - (bre[k] * bim[k2] - bim[k] * bre[k2]) * Hf[k][k2];
  float inv_o = 1.f / sqrtf(fmaxf(ss, EPSF));

  // ---- pass 2: o = sum_k beta_k * u_k (complex), normalized, in place ----
  for (int t = 0; t < DDIM / 32; ++t) {
    int dd = lane + 32 * t;
    float ore = 0.f, oim = 0.f;
#pragma unroll
    for (int k = 0; k < 4; ++k) {
      float ur = su[k][2 * dd], ui = su[k][2 * dd + 1];
      ore += bre[k] * ur - bim[k] * ui;
      oim += bre[k] * ui + bim[k] * ur;
    }
    psi[2 * dd]     = ore * inv_o;
    psi[2 * dd + 1] = oim * inv_o;
  }
}

// ---------------------------------------------------------------------------
// halt head + ACT update + merge + next pool. One wave per token.
// ---------------------------------------------------------------------------
__global__ void __launch_bounds__(256)
k_halt(const float* __restrict__ psi_h, const float* __restrict__ halt_w,
       const float* __restrict__ halt_b, const float* __restrict__ head_mix,
       float* __restrict__ cum, float* __restrict__ cost, int* __restrict__ halted,
       float* __restrict__ merged, unsigned short* __restrict__ pool, int is_last) {
  const int n = blockIdx.x * 8 + (threadIdx.x >> 5);
  const int lane = threadIdx.x & 31;

  float acc[12];
#pragma unroll
  for (int c = 0; c < 12; ++c) acc[c] = 0.f;
  float xv[32];
  for (int t = 0; t < 32; ++t) {
    int j = lane + 32 * t;
    float s = 0.25f * (psi_h[(size_t)(n * HN + 0) * KDIM + j] +
                       psi_h[(size_t)(n * HN + 1) * KDIM + j] +
                       psi_h[(size_t)(n * HN + 2) * KDIM + j] +
                       psi_h[(size_t)(n * HN + 3) * KDIM + j]);
    xv[t] = s;
#pragma unroll
    for (int c = 0; c < 12; ++c) acc[c] += s * halt_w[j * 12 + c];
  }
#pragma unroll
  for (int c = 0; c < 12; ++c) acc[c] = wsum(acc[c]) + halt_b[c];

  float wgt[4];
#pragma unroll
  for (int h = 0; h < 4; ++h) {
    float l0 = acc[3 * h], l1 = acc[3 * h + 1], l2 = acc[3 * h + 2];
    float m = fmaxf(l0, fmaxf(l1, l2));
    float e0 = __expf(l0 - m), e1 = __expf(l1 - m), e2 = __expf(l2 - m);
    float ph = (e0 + e1) / (e0 + e1 + e2);
    float cu_ = cum[n * HN + h];
    int   hd  = halted[n * HN + h];
    bool still = (hd == 0);
    float ncum = cu_ + ph;
    bool cross = ncum >= THRESHF;
    bool take  = cross || (is_last != 0);
    wgt[h] = still ? (take ? (1.f - cu_) : ph) : 0.f;
    if (lane == 0) {
      if (still) cost[n * HN + h] += ph;
      cum[n * HN + h]    = still ? (cross ? 1.f : ncum) : cu_;
      halted[n * HN + h] = (hd || (still && take)) ? 1 : 0;
    }
  }
  // head weights = softmax(head_mix)
  float hm0 = head_mix[0], hm1 = head_mix[1], hm2 = head_mix[2], hm3 = head_mix[3];
  float hmm = fmaxf(fmaxf(hm0, hm1), fmaxf(hm2, hm3));
  float he0 = __expf(hm0 - hmm), he1 = __expf(hm1 - hmm);
  float he2 = __expf(hm2 - hmm), he3 = __expf(hm3 - hmm);
  float hinv = 1.f / (he0 + he1 + he2 + he3);
  float wg[4] = {wgt[0] * he0 * hinv, wgt[1] * he1 * hinv,
                 wgt[2] * he2 * hinv, wgt[3] * he3 * hinv};

  for (int t = 0; t < 32; ++t) {
    int j = lane + 32 * t;
    float m = 0.f;
#pragma unroll
    for (int h = 0; h < 4; ++h) m += wg[h] * psi_h[(size_t)(n * HN + h) * KDIM + j];
    merged[(size_t)n * KDIM + j] += m;
    pool[(size_t)n * KDIM + j] = f2bf(xv[t]);   // next iteration's GEMM input
  }
}

// ---------------------------------------------------------------------------
// out = psi + out_scale * (merged - psi)
// ---------------------------------------------------------------------------
__global__ void __launch_bounds__(256)
k_output(const float* __restrict__ psi, const float* __restrict__ merged,
         const float* __restrict__ out_scale, float* __restrict__ out, int count) {
  int i = blockIdx.x * 256 + threadIdx.x;
  if (i < count) {
    float p = psi[i];
    out[i] = p + out_scale[0] * (merged[i] - p);
  }
}

__global__ void __launch_bounds__(256)
k_cost(const float* __restrict__ cost, float* __restrict__ out_scalar) {
  __shared__ float red[256];
  int tid = threadIdx.x;
  int i = blockIdx.x * 256 + tid;
  red[tid] = (i < NTOK * HN) ? cost[i] : 0.f;
  __syncthreads();
  for (int s = 128; s; s >>= 1) {
    if (tid < s) red[tid] += red[tid + s];
    __syncthreads();
  }
  if (tid == 0) atomicAdd(out_scalar, red[0] * (1.f / (float)(NTOK * HN)));
}

// ---------------------------------------------------------------------------
extern "C" void kernel_launch(void* const* d_in, const int* in_sizes, int n_in,
                              void* d_out, int out_size, void* d_ws, size_t ws_size,
                              hipStream_t stream) {
  (void)in_sizes; (void)n_in; (void)ws_size;
  const float* psi       = (const float*)d_in[0];
  const float* bank_keys = (const float*)d_in[1];
  const float* bank_u    = (const float*)d_in[2];
  const float* bank_v    = (const float*)d_in[3];
  const float* mix_u     = (const float*)d_in[4];
  const float* mix_v     = (const float*)d_in[5];
  const float* halt_w    = (const float*)d_in[6];
  const float* halt_b    = (const float*)d_in[7];
  const float* head_mix  = (const float*)d_in[8];
  const float* out_scale = (const float*)d_in[9];
  float* out = (float*)d_out;
  float* cost_out = out + (out_size - 1);

  char* ws = (char*)d_ws;
  float*          psi_h  = (float*)ws;                                      // 64 MiB
  unsigned short* poolb  = (unsigned short*)(ws + (((size_t)64) << 20));    //  8 MiB
  unsigned short* keysb  = (unsigned short*)(ws + (((size_t)72) << 20));    // 16 MiB
  float*          merged = (float*)(ws + (((size_t)88) << 20));             // 16 MiB
  int*            selidx = (int*)  (ws + (((size_t)104) << 20));
  float*          selw   = (float*)(ws + (((size_t)104) << 20) + (1u << 20));
  float*          cum    = (float*)(ws + (((size_t)104) << 20) + (2u << 20));
  float*          cost   = (float*)(ws + (((size_t)104) << 20) + (3u << 20));
  int*            halted = (int*)  (ws + (((size_t)104) << 20) + (4u << 20));

  // init
  k_init_keys<<<(HN * MBANK * KDIM + 255) / 256, 256, 0, stream>>>(
      bank_keys, keysb, HN * MBANK * KDIM);
  k_init_state<<<(NTOK * KDIM + 255) / 256, 256, 0, stream>>>(
      psi, psi_h, poolb, merged, cum, cost, halted, cost_out);

  // ACT iterations (fixed count, deterministic)
  for (int it = 0; it < TMAXI; ++it) {
    k_probe_topk<<<NTOK / 16, 256, PROBE_LDS, stream>>>(poolb, keysb, selidx, selw);
    k_apply<<<(NTOK * HN) / 8, 256, 0, stream>>>(
        psi_h, bank_u, bank_v, mix_u, mix_v, selidx, selw);
    k_halt<<<NTOK / 8, 256, 0, stream>>>(
        psi_h, halt_w, halt_b, head_mix, cum, cost, halted, merged, poolb,
        (it == TMAXI - 1) ? 1 : 0);
  }

  k_output<<<(NTOK * KDIM + 255) / 256, 256, 0, stream>>>(
      psi, merged, out_scale, out, out_size - 1);
  k_cost<<<(NTOK * HN + 255) / 256, 256, 0, stream>>>(cost, cost_out);
}
